// KBLaMBitNetMLP_18047452578029
// MI455X (gfx1250) — 8-bit matrix operand order repaired and checked by reading, compiled, not run on hardware
//
#include <hip/hip_runtime.h>


typedef __attribute__((ext_vector_type(8))) int v8i;

#define M_DIM 4096
#define H_DIM 4096
#define I_DIM 11008

#if __has_builtin(__builtin_amdgcn_global_load_async_to_lds_b128)
#define HAS_ASYNC_LDS 1
#else
#define HAS_ASYNC_LDS 0
#endif

typedef int v4i_raw __attribute__((vector_size(16)));
typedef __attribute__((address_space(1))) v4i_raw* gv4_t;
typedef __attribute__((address_space(3))) v4i_raw* lv4_t;

__device__ __forceinline__ void copy16_g2l(int* lds, const signed char* g) {
#if HAS_ASYNC_LDS
    __builtin_amdgcn_global_load_async_to_lds_b128((gv4_t)g, (lv4_t)lds, 0, 0);
#else
    int4 v = *reinterpret_cast<const int4*>(g);
    lds[0] = v.x; lds[1] = v.y; lds[2] = v.z; lds[3] = v.w;
#endif
}

__device__ __forceinline__ void wait_async_all() {
#if HAS_ASYNC_LDS
#if __has_builtin(__builtin_amdgcn_s_wait_asynccnt)
    __builtin_amdgcn_s_wait_asynccnt(0);
#else
    asm volatile("s_wait_asynccnt 0" ::: "memory");
#endif
#endif
}

__global__ __launch_bounds__(256)
void absmean_partial(const float* __restrict__ w, int n, float* __restrict__ partial) {
    __shared__ float red[256];
    float s = 0.0f;
    for (int i = blockIdx.x * blockDim.x + threadIdx.x; i < n; i += gridDim.x * blockDim.x)
        s += fabsf(w[i]);
    red[threadIdx.x] = s;
    __syncthreads();
    for (int off = 128; off > 0; off >>= 1) {
        if (threadIdx.x < off) red[threadIdx.x] += red[threadIdx.x + off];
        __syncthreads();
    }
    if (threadIdx.x == 0) partial[blockIdx.x] = red[0];
}

__global__ __launch_bounds__(256)
void absmean_final(const float* __restrict__ partial, int nPartial, int n,
                   float* __restrict__ scales) {
    __shared__ float red[256];
    float s = 0.0f;
    for (int i = threadIdx.x; i < nPartial; i += 256) s += partial[i];
    red[threadIdx.x] = s;
    __syncthreads();
    for (int off = 128; off > 0; off >>= 1) {
        if (threadIdx.x < off) red[threadIdx.x] += red[threadIdx.x + off];
        __syncthreads();
    }
    if (threadIdx.x == 0) {
        float mean = fmaxf(red[0] / (float)n, 1e-5f);
        scales[0] = 1.0f / mean;
        scales[1] = mean;
    }
}

__global__ __launch_bounds__(256)
void ternarize4(const float* __restrict__ w, signed char* __restrict__ wq,
                int n4, const float* __restrict__ scales) {
    int i = blockIdx.x * 256 + threadIdx.x;
    if (i >= n4) return;
    const float s = scales[0];
    float4 v = reinterpret_cast<const float4*>(w)[i];
    char4 o;
    o.x = (signed char)fminf(fmaxf(rintf(v.x * s), -1.0f), 1.0f);
    o.y = (signed char)fminf(fmaxf(rintf(v.y * s), -1.0f), 1.0f);
    o.z = (signed char)fminf(fmaxf(rintf(v.z * s), -1.0f), 1.0f);
    o.w = (signed char)fminf(fmaxf(rintf(v.w * s), -1.0f), 1.0f);
    reinterpret_cast<char4*>(wq)[i] = o;
}

__global__ __launch_bounds__(256)
void act_quant(const float* __restrict__ x, int K,
               signed char* __restrict__ xq, float* __restrict__ sinv) {
    const int row = blockIdx.x;
    const float* xr = x + (size_t)row * K;
    __shared__ float red[256];
    float m = 0.0f;
    for (int i = threadIdx.x; i < K; i += 256) m = fmaxf(m, fabsf(xr[i]));
    red[threadIdx.x] = m;
    __syncthreads();
    for (int off = 128; off > 0; off >>= 1) {
        if (threadIdx.x < off) red[threadIdx.x] = fmaxf(red[threadIdx.x], red[threadIdx.x + off]);
        __syncthreads();
    }
    const float amax  = fmaxf(red[0], 1e-5f);
    const float scale = 127.0f / amax;
    if (threadIdx.x == 0) sinv[row] = amax * (1.0f / 127.0f);
    signed char* out = xq + (size_t)row * K;
    for (int i = threadIdx.x; i < K; i += 256) {
        float v = fminf(fmaxf(rintf(xr[i] * scale), -128.0f), 127.0f);
        out[i] = (signed char)v;
    }
}

template <bool DUAL>
__global__ __launch_bounds__(256)
void bitgemm_wmma(const signed char* __restrict__ Aq,
                  const signed char* __restrict__ B0,
                  const signed char* __restrict__ B1,
                  const float* __restrict__ a_sinv,
                  const float* __restrict__ wsv0,
                  const float* __restrict__ wsv1,
                  float* __restrict__ out,
                  int M, int N, int K) {
    constexpr int LDW = 20;
    __shared__ int sA [2][128 * LDW];
    __shared__ int sB0[2][ 64 * LDW];
    __shared__ int sB1[2][DUAL ? 64 * LDW : 2];

    const int tid    = threadIdx.x;
    const int lane   = tid & 31;
    const int wv     = tid >> 5;
    const int waveM  = (wv & 3) * 32;
    const int waveN  = (wv >> 2) * 32;
    const int blockM = blockIdx.y * 128;
    const int blockN = blockIdx.x * 64;

    const v8i vzero = {};
    v8i accG[2][2], accU[2][2];
#pragma unroll
    for (int i = 0; i < 2; i++) {
        accG[i][0] = vzero; accG[i][1] = vzero;
        accU[i][0] = vzero; accU[i][1] = vzero;
    }

    const int brow = tid >> 2;
    const int bwc  = (tid & 3) * 4;

    auto load_tiles = [&](int buf, int k0) {
        for (int c = tid; c < 512; c += 256) {
            const int r  = c >> 2;
            const int wc = (c & 3) * 4;
            copy16_g2l(&sA[buf][r * LDW + wc],
                       Aq + (size_t)(blockM + r) * K + k0 + wc * 4);
        }
        copy16_g2l(&sB0[buf][brow * LDW + bwc],
                   B0 + (size_t)(blockN + brow) * K + k0 + bwc * 4);
        if constexpr (DUAL)
            copy16_g2l(&sB1[buf][brow * LDW + bwc],
                       B1 + (size_t)(blockN + brow) * K + k0 + bwc * 4);
    };

    auto compute_tile = [&](int buf) {
        const int half = lane >> 4;
        const int l15  = lane & 15;
        v8i af[2];
#pragma unroll
        for (int t = 0; t < 2; t++) {
            const int base = (waveM + t * 16 + l15) * LDW;
            const int h2 = half * 2;
#pragma unroll
            for (int p = 0; p < 4; p++) {
                af[t][2 * p]     = sA[buf][base + (p >> 1) * 8 + h2 * 2 + (p & 1) * 2];
                af[t][2 * p + 1] = sA[buf][base + (p >> 1) * 8 + h2 * 2 + (p & 1) * 2 + 1];
            }
        }
        const int kq = half * 4;
        v8i bf0[2], bf1[2];
#pragma unroll
        for (int t = 0; t < 2; t++) {
            const int base = (waveN + t * 16 + l15) * LDW;
#pragma unroll
            for (int j = 0; j < 4; j++) {
                bf0[t][j]     = sB0[buf][base + kq + j];
                bf0[t][4 + j] = sB0[buf][base + 8 + kq + j];
            }
            if constexpr (DUAL) {
#pragma unroll
                for (int j = 0; j < 4; j++) {
                    bf1[t][j]     = sB1[buf][base + kq + j];
                    bf1[t][4 + j] = sB1[buf][base + 8 + kq + j];
                }
            }
        }
#pragma unroll
        for (int i = 0; i < 2; i++) {
#pragma unroll
            for (int j = 0; j < 2; j++) {
                accG[i][j] = __builtin_amdgcn_wmma_i32_16x16x64_iu8(
                    true, af[i], true, bf0[j], accG[i][j], false, false);
                if constexpr (DUAL)
                    accU[i][j] = __builtin_amdgcn_wmma_i32_16x16x64_iu8(
                        true, af[i], true, bf1[j], accU[i][j], false, false);
            }
        }
    };

    load_tiles(0, 0);
    wait_async_all();
    __syncthreads();
    int cur = 0;
    for (int k0 = 64; k0 < K; k0 += 64) {
        load_tiles(cur ^ 1, k0);
        compute_tile(cur);
        wait_async_all();
        __syncthreads();
        cur ^= 1;
    }
    compute_tile(cur);

    const float w0 = *wsv0;
    const float w1 = DUAL ? *wsv1 : 0.0f;
    const int half = lane >> 4;
    const int nc   = lane & 15;
#pragma unroll
    for (int i = 0; i < 2; i++) {
#pragma unroll
        for (int r = 0; r < 8; r++) {
            const int grow = blockM + waveM + i * 16 + half * 8 + r;
            const float s = a_sinv[grow];
#pragma unroll
            for (int j = 0; j < 2; j++) {
                const int gcol = blockN + waveN + j * 16 + nc;
                const size_t o = (size_t)grow * N + gcol;
                const float g = (float)accG[i][j][r] * (s * w0);
                if constexpr (DUAL) {
                    const float u  = (float)accU[i][j][r] * (s * w1);
                    const float rg = fmaxf(g, 0.0f);
                    out[o] = rg * rg * u;
                } else {
                    out[o] = g;
                }
            }
        }
    }
}

extern "C" void kernel_launch(void* const* d_in, const int* in_sizes, int n_in,
                              void* d_out, int out_size, void* d_ws, size_t ws_size,
                              hipStream_t stream) {
    (void)in_sizes; (void)n_in; (void)out_size; (void)ws_size;
    const float* x  = (const float*)d_in[0];
    const float* wg = (const float*)d_in[1];
    const float* wu = (const float*)d_in[2];
    const float* wd = (const float*)d_in[3];
    float* out = (float*)d_out;

    char* p = (char*)d_ws;
    auto take = [&](size_t bytes) -> char* {
        char* q = p;
        p += (bytes + 255) & ~(size_t)255;
        return q;
    };

    signed char* xq   = (signed char*)take((size_t)M_DIM * H_DIM);
    signed char* wgq  = (signed char*)take((size_t)I_DIM * H_DIM);
    signed char* wuq  = (signed char*)take((size_t)I_DIM * H_DIM);
    signed char* wdq  = (signed char*)take((size_t)H_DIM * I_DIM);
    signed char* gq   = (signed char*)take((size_t)M_DIM * I_DIM);
    float* gated      = (float*)take((size_t)M_DIM * I_DIM * sizeof(float));
    float* x_sinv     = (float*)take(M_DIM * sizeof(float));
    float* g_sinv     = (float*)take(M_DIM * sizeof(float));
    float* sc_g       = (float*)take(2 * sizeof(float));
    float* sc_u       = (float*)take(2 * sizeof(float));
    float* sc_d       = (float*)take(2 * sizeof(float));
    float* part_g     = (float*)take(512 * sizeof(float));
    float* part_u     = (float*)take(512 * sizeof(float));
    float* part_d     = (float*)take(512 * sizeof(float));

    const int nW = I_DIM * H_DIM;

    absmean_partial<<<512, 256, 0, stream>>>(wg, nW, part_g);
    absmean_partial<<<512, 256, 0, stream>>>(wu, nW, part_u);
    absmean_partial<<<512, 256, 0, stream>>>(wd, nW, part_d);
    absmean_final<<<1, 256, 0, stream>>>(part_g, 512, nW, sc_g);
    absmean_final<<<1, 256, 0, stream>>>(part_u, 512, nW, sc_u);
    absmean_final<<<1, 256, 0, stream>>>(part_d, 512, nW, sc_d);

    const int n4 = nW / 4;
    const int tg = (n4 + 255) / 256;
    ternarize4<<<tg, 256, 0, stream>>>(wg, wgq, n4, sc_g);
    ternarize4<<<tg, 256, 0, stream>>>(wu, wuq, n4, sc_u);
    ternarize4<<<tg, 256, 0, stream>>>(wd, wdq, n4, sc_d);

    act_quant<<<M_DIM, 256, 0, stream>>>(x, H_DIM, xq, x_sinv);

    dim3 g1(I_DIM / 64, M_DIM / 128);
    bitgemm_wmma<true><<<g1, 256, 0, stream>>>(xq, wgq, wuq, x_sinv,
                                               sc_g + 1, sc_u + 1, gated,
                                               M_DIM, I_DIM, H_DIM);

    act_quant<<<M_DIM, 256, 0, stream>>>(gated, I_DIM, gq, g_sinv);

    dim3 g2(H_DIM / 64, M_DIM / 128);
    bitgemm_wmma<false><<<g2, 256, 0, stream>>>(gq, wdq, nullptr, g_sinv,
                                                sc_d + 1, nullptr, out,
                                                M_DIM, H_DIM, I_DIM);
}
